// MultiHeadAttention_67860483277320
// MI455X (gfx1250) — compile-verified, matrix operand order checked by reading, not run on hardware
//
#include <hip/hip_runtime.h>

typedef __attribute__((ext_vector_type(16))) _Float16 v16h;
typedef __attribute__((ext_vector_type(8)))  _Float16 v8h;
typedef __attribute__((ext_vector_type(8)))  float    v8f;
typedef __attribute__((ext_vector_type(4)))  float    f4;

#define N_SEQ 2048
#define DM    1024
#define NH    16
#define DK    64
#define SL    2052   // padded LDS row stride (floats) for score rows

static __device__ __forceinline__ v8f wmma16(v16h a, v16h b, v8f c) {
  return __builtin_amdgcn_wmma_f32_16x16x32_f16(false, a, false, b, (short)0, c,
                                                false, false);
}

// Load a 16x32 f16 WMMA fragment (A or B operand) from row-major storage.
// ISA layout: lane L holds row M=(L&15); element e maps to
// K = 8*(L>>4) + e        for e<8
// K = 16 + 8*(L>>4) + e-8 for e>=8   -> two contiguous v8h (b128) loads.
static __device__ __forceinline__ v16h frag_h(const _Float16* p, int stride_h) {
  const int lane = threadIdx.x & 31;
  const _Float16* r = p + (size_t)(lane & 15) * stride_h + ((lane >> 4) * 8);
  v8h lo = *(const v8h*)(r);
  v8h hi = *(const v8h*)(r + 16);
  v16h out;
#pragma unroll
  for (int i = 0; i < 8; ++i) { out[i] = lo[i]; out[i + 8] = hi[i]; }
  return out;
}

// Same fragment but sourced from f32 storage (converted to f16 in regs).
static __device__ __forceinline__ v16h frag_f(const float* p, int stride_f) {
  const int lane = threadIdx.x & 31;
  const float* r = p + (size_t)(lane & 15) * stride_f + ((lane >> 4) * 8);
  f4 a0 = *(const f4*)(r);
  f4 a1 = *(const f4*)(r + 4);
  f4 a2 = *(const f4*)(r + 16);
  f4 a3 = *(const f4*)(r + 20);
  v16h out;
#pragma unroll
  for (int i = 0; i < 4; ++i) {
    out[i]      = (_Float16)a0[i];
    out[i + 4]  = (_Float16)a1[i];
    out[i + 8]  = (_Float16)a2[i];
    out[i + 12] = (_Float16)a3[i];
  }
  return out;
}

// Generic GEMM: C(MxN) = A(MxK) @ B(KxN), epilogue (acc + bias[n]) * scale.
// A_F32: A operand is f32 (converted to f16 while staging) vs already f16.
// OUT_MODE: 0 -> f16 row-major (ld=N), 1 -> f16 transposed (out[n*M+m],
// used to build V^T), 2 -> f32 row-major.
template <bool A_F32, int OUT_MODE>
__global__ __launch_bounds__(256) void gemm_kernel(
    const void* __restrict__ Av, const float* __restrict__ B,
    const float* __restrict__ bias, _Float16* __restrict__ outH,
    float* __restrict__ outF, int M, int N, int K, float scale) {
  __shared__ _Float16 sA[128 * 40];   // 128 rows x 32 K, stride 40 (pad)
  __shared__ _Float16 sB[128 * 40];   // 128 cols x 32 K (transposed), stride 40

  const int tid = threadIdx.x;
  const int lane = tid & 31;
  const int wid = tid >> 5;
  const int wm = wid & 3;    // 4 row groups of 32
  const int wn = wid >> 2;   // 2 col groups of 64
  const int n0 = blockIdx.x * 128;
  const int m0 = blockIdx.y * 128;

  // per-thread staging coordinates
  const int a_row = tid >> 1;          // 0..127
  const int a_kb  = (tid & 1) * 16;    // 0 or 16
  const int b_kr  = tid >> 3;          // 0..31
  const int b_nb  = (tid & 7) * 16;    // 0..112

  v8f acc[2][4];
#pragma unroll
  for (int i = 0; i < 2; ++i)
#pragma unroll
    for (int j = 0; j < 4; ++j)
      acc[i][j] = (v8f){0.f, 0.f, 0.f, 0.f, 0.f, 0.f, 0.f, 0.f};

  for (int k0 = 0; k0 < K; k0 += 32) {
    // --- stage A tile: 128x32, each thread covers one half-row (16 elems) ---
    {
      _Float16* dst = sA + a_row * 40 + a_kb;
      if (A_F32) {
        const float* src =
            (const float*)Av + (size_t)(m0 + a_row) * K + k0 + a_kb;
#pragma unroll
        for (int i = 0; i < 16; i += 4) {
          f4 v = *(const f4*)(src + i);
          dst[i + 0] = (_Float16)v[0];
          dst[i + 1] = (_Float16)v[1];
          dst[i + 2] = (_Float16)v[2];
          dst[i + 3] = (_Float16)v[3];
        }
        if (k0 + 32 < K) __builtin_prefetch(src + 32, 0, 1);
      } else {
        const _Float16* src =
            (const _Float16*)Av + (size_t)(m0 + a_row) * K + k0 + a_kb;
        *(v8h*)(dst)     = *(const v8h*)(src);
        *(v8h*)(dst + 8) = *(const v8h*)(src + 8);
        if (k0 + 32 < K) __builtin_prefetch(src + 32, 0, 1);
      }
    }
    // --- stage B tile transposed into [n][k]: thread reads 16 contiguous n ---
    {
      const float* src = B + (size_t)(k0 + b_kr) * N + n0 + b_nb;
#pragma unroll
      for (int i = 0; i < 16; i += 4) {
        f4 v = *(const f4*)(src + i);
        sB[(b_nb + i + 0) * 40 + b_kr] = (_Float16)v[0];
        sB[(b_nb + i + 1) * 40 + b_kr] = (_Float16)v[1];
        sB[(b_nb + i + 2) * 40 + b_kr] = (_Float16)v[2];
        sB[(b_nb + i + 3) * 40 + b_kr] = (_Float16)v[3];
      }
      if (k0 + 32 < K) __builtin_prefetch(src + (size_t)32 * N, 0, 1);
    }
    __syncthreads();

    v16h aF[2], bF[4];
#pragma unroll
    for (int t = 0; t < 2; ++t) aF[t] = frag_h(sA + (wm * 32 + t * 16) * 40, 40);
#pragma unroll
    for (int t = 0; t < 4; ++t) bF[t] = frag_h(sB + (wn * 64 + t * 16) * 40, 40);
#pragma unroll
    for (int t = 0; t < 2; ++t)
#pragma unroll
      for (int u = 0; u < 4; ++u) acc[t][u] = wmma16(aF[t], bF[u], acc[t][u]);
    __syncthreads();
  }

  // --- epilogue: C layout VGPR r -> M = r + 8*(lane>=16), N = lane&15 ---
#pragma unroll
  for (int t = 0; t < 2; ++t)
#pragma unroll
    for (int u = 0; u < 4; ++u)
#pragma unroll
      for (int r = 0; r < 8; ++r) {
        const int gm = m0 + wm * 32 + t * 16 + r + (lane >> 4) * 8;
        const int gn = n0 + wn * 64 + u * 16 + (lane & 15);
        const float v = (acc[t][u][r] + bias[gn]) * scale;
        if (OUT_MODE == 0)      outH[(size_t)gm * N + gn] = (_Float16)v;
        else if (OUT_MODE == 1) outH[(size_t)gn * M + gm] = (_Float16)v;
        else                    outF[(size_t)gm * N + gn] = v;
      }
}

// One block = one head x 32 query rows. Scores live in 256.5KB dynamic LDS
// (CDNA5 WGP has 320KB): compute scores with WMMA, softmax in place, stream
// the attn matrix (the bandwidth-dominant 268MB output) once, then P@V.
__global__ __launch_bounds__(256) void attn_kernel(
    const _Float16* __restrict__ qh, const _Float16* __restrict__ kh,
    const _Float16* __restrict__ vt, float* __restrict__ attn,
    _Float16* __restrict__ xh) {
  extern __shared__ float sS[];        // [32][SL]
  const int tid = threadIdx.x;
  const int lane = tid & 31;
  const int w = tid >> 5;              // wave id 0..7
  const int n0 = blockIdx.x * 32;
  const int h = blockIdx.y;

  // ---- Phase 1: S = (Q/8) @ K^T for 32 rows x 2048 cols ----
  v16h aF[2][2];
#pragma unroll
  for (int t = 0; t < 2; ++t)
#pragma unroll
    for (int ks = 0; ks < 2; ++ks)
      aF[t][ks] = frag_h(qh + (size_t)(n0 + t * 16) * DM + h * DK + ks * 32, DM);

  for (int ct = 0; ct < 16; ++ct) {    // each wave owns 256 cols = 16 tiles
    const int col0 = w * 256 + ct * 16;
    v16h b0 = frag_h(kh + (size_t)col0 * DM + h * DK, DM);
    v16h b1 = frag_h(kh + (size_t)col0 * DM + h * DK + 32, DM);
    v8f s0 = (v8f){0.f, 0.f, 0.f, 0.f, 0.f, 0.f, 0.f, 0.f};
    v8f s1 = s0;
    s0 = wmma16(aF[0][0], b0, s0);  s0 = wmma16(aF[0][1], b1, s0);
    s1 = wmma16(aF[1][0], b0, s1);  s1 = wmma16(aF[1][1], b1, s1);
#pragma unroll
    for (int r = 0; r < 8; ++r) {
      const int rr = r + (lane >> 4) * 8;
      const int cc = col0 + (lane & 15);
      sS[rr * SL + cc]        = s0[r];
      sS[(16 + rr) * SL + cc] = s1[r];
    }
  }
  __syncthreads();

  // ---- Phase 2: row softmax + stream attn to HBM (f32) ----
#pragma unroll
  for (int q = 0; q < 4; ++q) {
    const int r = w * 4 + q;
    float* rp = sS + r * SL;
    float mx = -3.4e38f;
    for (int i = lane; i < N_SEQ; i += 32) mx = fmaxf(mx, rp[i]);
#pragma unroll
    for (int off = 16; off; off >>= 1) mx = fmaxf(mx, __shfl_xor(mx, off, 32));
    float sum = 0.f;
    for (int i = lane; i < N_SEQ; i += 32) {
      float e = __expf(rp[i] - mx);
      rp[i] = e;
      sum += e;
    }
#pragma unroll
    for (int off = 16; off; off >>= 1) sum += __shfl_xor(sum, off, 32);
    const float inv = 1.0f / sum;
    float* ap = attn + ((size_t)h * N_SEQ + (n0 + r)) * N_SEQ;
    for (int i = lane; i < N_SEQ; i += 32) {
      const float p = rp[i] * inv;
      rp[i] = p;
      ap[i] = p;
    }
  }
  __syncthreads();

  // ---- Phase 3: X = P @ V  (one 16x16 tile per wave; 32x64 total) ----
  const int tm = w >> 2, tn = w & 3;
  v8f acc = (v8f){0.f, 0.f, 0.f, 0.f, 0.f, 0.f, 0.f, 0.f};
  const _Float16* vbase = vt + (size_t)(h * DK + tn * 16) * N_SEQ;
  for (int k0 = 0; k0 < N_SEQ; k0 += 32) {
    v16h a = frag_f(sS + (size_t)(tm * 16) * SL + k0, SL);
    v16h b = frag_h(vbase + k0, N_SEQ);
    acc = wmma16(a, b, acc);
  }
#pragma unroll
  for (int r = 0; r < 8; ++r) {
    const int gm = n0 + tm * 16 + r + (lane >> 4) * 8;
    const int gc = h * DK + tn * 16 + (lane & 15);
    xh[(size_t)gm * DM + gc] = (_Float16)acc[r];
  }
}

extern "C" void kernel_launch(void* const* d_in, const int* in_sizes, int n_in,
                              void* d_out, int out_size, void* d_ws, size_t ws_size,
                              hipStream_t stream) {
  (void)in_sizes; (void)n_in; (void)out_size; (void)ws_size;
  const float* query = (const float*)d_in[0];
  const float* key_  = (const float*)d_in[1];
  const float* value = (const float*)d_in[2];
  const float* Wq = (const float*)d_in[3];
  const float* bq = (const float*)d_in[4];
  const float* Wk = (const float*)d_in[5];
  const float* bk = (const float*)d_in[6];
  const float* Wv = (const float*)d_in[7];
  const float* bv = (const float*)d_in[8];
  const float* Wo = (const float*)d_in[9];
  const float* bo = (const float*)d_in[10];

  float* out  = (float*)d_out;                    // (2048,1024)
  float* attn = out + (size_t)N_SEQ * DM;         // (16,2048,2048)

  _Float16* qh = (_Float16*)d_ws;                 // (2048,1024) f16
  _Float16* kh = qh + (size_t)N_SEQ * DM;         // (2048,1024) f16
  _Float16* vt = kh + (size_t)N_SEQ * DM;         // (1024,2048) f16 = V^T
  _Float16* xh = vt + (size_t)N_SEQ * DM;         // (2048,1024) f16

  const dim3 gg(DM / 128, N_SEQ / 128);
  // Q = (query@Wq + bq) / sqrt(d_k)   (scale folded here, exact *0.125)
  gemm_kernel<true, 0><<<gg, 256, 0, stream>>>(query, Wq, bq, qh, nullptr,
                                               N_SEQ, DM, DM, 0.125f);
  gemm_kernel<true, 0><<<gg, 256, 0, stream>>>(key_, Wk, bk, kh, nullptr,
                                               N_SEQ, DM, DM, 1.0f);
  gemm_kernel<true, 1><<<gg, 256, 0, stream>>>(value, Wv, bv, vt, nullptr,
                                               N_SEQ, DM, DM, 1.0f);
  attn_kernel<<<dim3(N_SEQ / 32, NH), 256, 32 * SL * sizeof(float), stream>>>(
      qh, kh, vt, attn, xh);
  gemm_kernel<false, 2><<<gg, 256, 0, stream>>>(xh, Wo, bo, nullptr, out,
                                                N_SEQ, DM, DM, 1.0f);
}
